// PetriNetAlignmentPredictor_2585570312336
// MI455X (gfx1250) — matrix operand order repaired and checked by reading, not compiled, not run on hardware
//
#include <hip/hip_runtime.h>

typedef __bf16 bf16_t;
typedef __bf16 v16bf __attribute__((ext_vector_type(16)));
typedef float  v8f   __attribute__((ext_vector_type(8)));
typedef int    v4i_t __attribute__((ext_vector_type(4)));

#define NM    50000   // markings
#define NN    64      // embedding dim
#define BB    256     // batch
#define NT    4096    // transitions
#define NGEN  2016    // generators
#define NSTEP 50
#define INDIM 128     // 2*NN
#define H1DIM 256
#define H2DIM 128

#define LDA   136     // padded LDS row stride for 128-wide rows (bank spread, 16B aligned)
#define LDE   72      // padded LDS row stride for 64-wide rows

#define AS1 __attribute__((address_space(1)))
#define AS3 __attribute__((address_space(3)))

// 16-byte global -> LDS copy. CDNA5 async-to-LDS path (ASYNCcnt).
static __device__ __forceinline__ void cp16(void* l, const void* g) {
#if __has_builtin(__builtin_amdgcn_global_load_async_to_lds_b128)
    __builtin_amdgcn_global_load_async_to_lds_b128((AS1 v4i_t*)g, (AS3 v4i_t*)l, 0, 0);
#else
    *(int4*)l = *(const int4*)g;
#endif
}
static __device__ __forceinline__ void cp_wait() {
#if __has_builtin(__builtin_amdgcn_s_wait_asynccnt)
    __builtin_amdgcn_s_wait_asynccnt(0);
#endif
}

// ---------------- embeddings: l2-normalize rows, emit f32 + bf16 ----------------
__global__ void k_norm_embs(const float* __restrict__ w, float* __restrict__ en,
                            bf16_t* __restrict__ eb) {
    int m = blockIdx.x * blockDim.x + threadIdx.x;
    if (m >= NM) return;
    const float* r = w + (size_t)m * NN;
    float ss = 0.f;
    #pragma unroll
    for (int j = 0; j < NN; ++j) { float v = r[j]; ss += v * v; }
    float inv = 1.0f / fmaxf(sqrtf(ss), 1e-12f);
    float*  o  = en + (size_t)m * NN;
    bf16_t* ob = eb + (size_t)m * NN;
    #pragma unroll
    for (int j = 0; j < NN; ++j) { float v = r[j] * inv; o[j] = v; ob[j] = (bf16_t)v; }
}

// ---------------- generic f32 -> bf16 convert ----------------
__global__ void k_cvt_bf16(const float* __restrict__ s, bf16_t* __restrict__ d, int n) {
    int i = blockIdx.x * blockDim.x + threadIdx.x;
    if (i < n) d[i] = (bf16_t)s[i];
}

// ---------------- W3 [128,4096] -> transposed bf16 w3t [4096,128] ----------------
__global__ void k_cvt_w3t(const float* __restrict__ W3, bf16_t* __restrict__ w3t) {
    int idx = blockIdx.x * 256 + threadIdx.x;   // over NT*INDIM
    int n = idx & (NT - 1);
    int k = idx >> 12;
    w3t[(size_t)n * INDIM + k] = (bf16_t)W3[(size_t)k * NT + n];
}

// ---------------- one-hot index extraction + embedding gather ----------------
__global__ void k_extract(const float* __restrict__ vsrc, const float* __restrict__ vtgt,
                          const float* __restrict__ en,
                          float* __restrict__ vcur, float* __restrict__ vtg) {
    int b = blockIdx.x, tid = threadIdx.x;
    __shared__ int sh[256];
    int best = -1;
    for (int m = tid; m < NM; m += 256)
        if (vsrc[(size_t)b * NM + m] > 0.5f) best = m;
    sh[tid] = best; __syncthreads();
    for (int s = 128; s > 0; s >>= 1) { if (tid < s) sh[tid] = max(sh[tid], sh[tid + s]); __syncthreads(); }
    int si = sh[0]; __syncthreads();
    best = -1;
    for (int m = tid; m < NM; m += 256)
        if (vtgt[(size_t)b * NM + m] > 0.5f) best = m;
    sh[tid] = best; __syncthreads();
    for (int s = 128; s > 0; s >>= 1) { if (tid < s) sh[tid] = max(sh[tid], sh[tid + s]); __syncthreads(); }
    int ti = sh[0];
    if (tid < NN) {
        vcur[b * NN + tid] = en[(size_t)si * NN + tid];
        vtg [b * NN + tid] = en[(size_t)ti * NN + tid];
    }
}

// ---------------- fused MLP layers 1+2 (small, f32 VALU, LDS staged) ----------------
__global__ void k_mlp12(const float* __restrict__ vcur, const float* __restrict__ vtg,
                        const float* __restrict__ W1, const float* __restrict__ b1,
                        const float* __restrict__ W2, const float* __restrict__ b2,
                        bf16_t* __restrict__ h2bf) {
    int b = blockIdx.x, t = threadIdx.x;   // 256 threads
    __shared__ float x[INDIM];
    __shared__ float h1[H1DIM];
    if (t < INDIM) x[t] = (t < NN) ? vcur[b * NN + t] : vtg[b * NN + (t - NN)];
    __syncthreads();
    float acc = b1[t];
    for (int k = 0; k < INDIM; ++k) acc += x[k] * W1[k * H1DIM + t];
    h1[t] = fmaxf(acc, 0.f);
    __syncthreads();
    if (t < H2DIM) {
        float a2 = b2[t];
        for (int k = 0; k < H1DIM; ++k) a2 += h1[k] * W2[k * H2DIM + t];
        h2bf[b * H2DIM + t] = (bf16_t)fmaxf(a2, 0.f);
    }
}

// ---------------- layer 3: [256,128] @ [128,4096] via WMMA bf16, LDS-staged ----------------
// Block = 256 threads (8 waves): tile 16(M) x 128(N), full K=128 staged in LDS via
// async-to-LDS b128 copies. B staged from pre-transposed w3t so both A and B fragment
// reads are contiguous ds_load_b128. Padded LDS strides avoid bank collisions.
// Writes logits (+bias) straight into logits_seq[b, step, t] output region.
__global__ void k_gemm3(const bf16_t* __restrict__ A, const bf16_t* __restrict__ Bt,
                        const float* __restrict__ bias, float* __restrict__ out, int step) {
    __shared__ bf16_t sA[16 * LDA];          //  ~4.3 KB
    __shared__ bf16_t sB[128 * LDA];         //  ~34 KB  (n-major: sB[n][k])
    int t   = threadIdx.x;
    int tm  = blockIdx.x >> 5;               // 0..15  (M tile)
    int tnb = blockIdx.x & 31;               // 0..31  (128-wide N chunk)

    {   // stage A tile 16x128 (256 x 16B)
        int row = t >> 4, ch = t & 15;
        cp16(&sA[row * LDA + ch * 8], A + (tm * 16 + row) * INDIM + ch * 8);
    }
    #pragma unroll
    for (int it = 0; it < 8; ++it) {         // stage B tile 128(n) x 128(k) (2048 x 16B)
        int idx = t + it * 256;
        int row = idx >> 4, ch = idx & 15;   // row = n-local
        cp16(&sB[row * LDA + ch * 8], Bt + (size_t)(tnb * 128 + row) * INDIM + ch * 8);
    }
    cp_wait();
    __syncthreads();

    int wave = t >> 5, lane = t & 31;
    int g = lane >> 4, c = lane & 15;
    int n0 = wave * 16;
    v8f acc = {};
    #pragma unroll
    for (int kk = 0; kk < INDIM; kk += 32) {
        v16bf a, bm;
        #pragma unroll
        for (int e = 0; e < 16; ++e) {
            int ka = g * 16 + e;                                         // A 16x32 layout
            a[e] = sA[c * LDA + kk + ka];
            int kb = g * 16 + e;                                         // B 32x16 layout
            bm[e] = sB[(n0 + c) * LDA + kk + kb];                        // contiguous in e
        }
        acc = __builtin_amdgcn_wmma_f32_16x16x32_bf16(false, a, false, bm,
                                                      (short)0, acc, false, false);
    }
    #pragma unroll
    for (int e = 0; e < 8; ++e) {            // C/D: VGPR e -> M = e + 8*g, N = lane%16
        int m = tm * 16 + e + 8 * g;
        int n = tnb * 128 + n0 + c;
        out[((size_t)m * NSTEP + step) * NT + n] = acc[e] + bias[n];
    }
}

// ---------------- argmax over logits + gumbel (first-max tiebreak) ----------------
__global__ void k_argmax(const float* __restrict__ logits, const float* __restrict__ gum,
                         int step, int* __restrict__ tsel) {
    int b = blockIdx.x, tid = threadIdx.x;   // 256 threads
    __shared__ float sv[256];
    __shared__ int   sidx[256];
    float bv = -3.4e38f; int bi = NT;
    const float* lr = logits + ((size_t)b * NSTEP + step) * NT;
    const float* gr = gum + ((size_t)step * BB + b) * NT;
    for (int t = tid; t < NT; t += 256) {
        float v = lr[t] + gr[t];
        if (v > bv) { bv = v; bi = t; }
    }
    sv[tid] = bv; sidx[tid] = bi; __syncthreads();
    for (int s = 128; s > 0; s >>= 1) {
        if (tid < s) {
            if (sv[tid + s] > sv[tid] ||
               (sv[tid + s] == sv[tid] && sidx[tid + s] < sidx[tid])) {
                sv[tid] = sv[tid + s]; sidx[tid] = sidx[tid + s];
            }
        }
        __syncthreads();
    }
    if (tid == 0) tsel[b] = sidx[0];
}

// ---------------- build skew-symmetric omega from selected transition ----------------
__global__ void k_omega(const float* __restrict__ coeffs, const int* __restrict__ tsel,
                        float* __restrict__ om) {
    int b = blockIdx.x, tid = threadIdx.x;   // 256 threads
    float* ob = om + (size_t)b * NN * NN;
    for (int i = tid; i < NN * NN; i += 256) ob[i] = 0.f;
    __syncthreads();
    const float* cr = coeffs + (size_t)tsel[b] * NGEN;
    for (int k = tid; k < NGEN; k += 256) {
        int rem = k, i = 0;
        while (rem >= NN - 1 - i) { rem -= NN - 1 - i; ++i; }   // triu row-major
        int j = i + 1 + rem;
        float cv = cr[k];
        ob[i * NN + j] = -cv;
        ob[j * NN + i] =  cv;
    }
}

// ---------------- v <- normalize(expm(omega) @ v) via Taylor-on-vector ----------------
__global__ void k_rotate(const float* __restrict__ om, float* __restrict__ vcur) {
    int b = blockIdx.x, i = threadIdx.x;     // 64 threads (2 waves)
    __shared__ float tl[NN];
    __shared__ float red[NN];
    float w[NN];
    const float* orow = om + (size_t)b * NN * NN + i * NN;
    #pragma unroll
    for (int j = 0; j < NN; ++j) w[j] = orow[j];
    float x = vcur[b * NN + i];
    tl[i] = x; float acc = x;
    __syncthreads();
    for (int k = 1; k <= 48; ++k) {          // ||omega|| ~ 6 -> 48 terms ample for fp32
        float s = 0.f;
        #pragma unroll
        for (int j = 0; j < NN; ++j) s += w[j] * tl[j];
        s *= (1.0f / (float)k);
        __syncthreads();
        tl[i] = s; acc += s;
        __syncthreads();
    }
    red[i] = acc * acc; __syncthreads();
    for (int s2 = 32; s2 > 0; s2 >>= 1) { if (i < s2) red[i] += red[i + s2]; __syncthreads(); }
    float denom = fmaxf(sqrtf(red[0]), 1e-12f);
    vcur[b * NN + i] = acc / denom;
}

// ---------------- final scoring: [256,64] @ embs^T [64,50000] via WMMA bf16 ----------------
// Block = 256 threads (8 waves): tile 16(M) x 128(N), K=64 staged in LDS.
// Embedding tile staged row-major per marking (B[k,n] = E[n][k]), padded stride.
__global__ void k_score(const bf16_t* __restrict__ A, const bf16_t* __restrict__ E,
                        float* __restrict__ out) {
    __shared__ bf16_t sA[16 * LDE];          //  ~2.3 KB
    __shared__ bf16_t sE[128 * LDE];         //  ~18 KB
    int t   = threadIdx.x;
    int tm  = blockIdx.x / 391;              // 0..15
    int tnb = blockIdx.x % 391;              // 0..390 (128-wide N chunk; last partial)

    if (t < 128) {                           // stage A tile 16x64 (128 x 16B)
        int row = t >> 3, ch = t & 7;
        cp16(&sA[row * LDE + ch * 8], A + (tm * 16 + row) * NN + ch * 8);
    }
    #pragma unroll
    for (int it = 0; it < 4; ++it) {         // stage E tile 128x64 (1024 x 16B)
        int idx = t + it * 256;
        int row = idx >> 3, ch = idx & 7;    // row-local marking, 8-elem chunk
        cp16(&sE[row * LDE + ch * 8], E + (size_t)(tnb * 128 + row) * NN + ch * 8);
    }
    cp_wait();
    __syncthreads();

    int wave = t >> 5, lane = t & 31;
    int g = lane >> 4, c = lane & 15;
    int n0 = wave * 16;
    v8f acc = {};
    #pragma unroll
    for (int kk = 0; kk < NN; kk += 32) {
        v16bf a, bm;
        #pragma unroll
        for (int e = 0; e < 16; ++e) {
            int ka = g * 16 + e;
            a[e] = sA[c * LDE + kk + ka];
            int kb = g * 16 + e;
            bm[e] = sE[(n0 + c) * LDE + kk + kb];   // B[k,n] = E[n][k], contiguous in e
        }
        acc = __builtin_amdgcn_wmma_f32_16x16x32_bf16(false, a, false, bm,
                                                      (short)0, acc, false, false);
    }
    #pragma unroll
    for (int e = 0; e < 8; ++e) {
        int m = tm * 16 + e + 8 * g;
        int n = tnb * 128 + n0 + c;
        if (n < NM) out[(size_t)m * NM + n] = acc[e];
    }
}

// ---------------- in-place row softmax over 50000 ----------------
__global__ void k_softmax(float* __restrict__ out) {
    int b = blockIdx.x, tid = threadIdx.x;   // 1024 threads
    float* row = out + (size_t)b * NM;
    __shared__ float sh[1024];
    float mx = -3.4e38f;
    for (int m = tid; m < NM; m += 1024) mx = fmaxf(mx, row[m]);
    sh[tid] = mx; __syncthreads();
    for (int s = 512; s > 0; s >>= 1) { if (tid < s) sh[tid] = fmaxf(sh[tid], sh[tid + s]); __syncthreads(); }
    mx = sh[0]; __syncthreads();
    float sum = 0.f;
    for (int m = tid; m < NM; m += 1024) sum += __expf(row[m] - mx);
    sh[tid] = sum; __syncthreads();
    for (int s = 512; s > 0; s >>= 1) { if (tid < s) sh[tid] += sh[tid + s]; __syncthreads(); }
    float inv = 1.0f / sh[0];
    for (int m = tid; m < NM; m += 1024) row[m] = __expf(row[m] - mx) * inv;
}

extern "C" void kernel_launch(void* const* d_in, const int* in_sizes, int n_in,
                              void* d_out, int out_size, void* d_ws, size_t ws_size,
                              hipStream_t stream) {
    (void)in_sizes; (void)n_in; (void)out_size; (void)ws_size;
    const float* v_src = (const float*)d_in[0];
    const float* v_tgt = (const float*)d_in[1];
    const float* embw  = (const float*)d_in[2];
    const float* coeff = (const float*)d_in[3];
    const float* W1 = (const float*)d_in[4];
    const float* b1 = (const float*)d_in[5];
    const float* W2 = (const float*)d_in[6];
    const float* b2 = (const float*)d_in[7];
    const float* W3 = (const float*)d_in[8];
    const float* b3 = (const float*)d_in[9];
    const float* gum = (const float*)d_in[10];

    float* out    = (float*)d_out;
    float* pred   = out;                          // [256, 50000]
    float* logits = out + (size_t)BB * NM;        // [256, 50, 4096]

    char* ws = (char*)d_ws;
    size_t off = 0;
    auto alloc = [&](size_t bytes) -> void* {
        void* p = ws + off;
        off += (bytes + 255) & ~((size_t)255);
        return p;
    };
    float*  embs_n = (float*) alloc((size_t)NM * NN * 4);       // 12.8 MB
    bf16_t* embs_b = (bf16_t*)alloc((size_t)NM * NN * 2);       //  6.4 MB
    bf16_t* w3t    = (bf16_t*)alloc((size_t)NT * INDIM * 2);    //  1.0 MB (transposed)
    float*  vcur   = (float*) alloc((size_t)BB * NN * 4);
    float*  vtg    = (float*) alloc((size_t)BB * NN * 4);
    bf16_t* h2b    = (bf16_t*)alloc((size_t)BB * H2DIM * 2);
    int*    tsel   = (int*)   alloc((size_t)BB * 4);
    float*  omg    = (float*) alloc((size_t)BB * NN * NN * 4);  //  4.0 MB
    bf16_t* vfb    = (bf16_t*)alloc((size_t)BB * NN * 2);

    k_norm_embs<<<(NM + 255) / 256, 256, 0, stream>>>(embw, embs_n, embs_b);
    k_cvt_w3t<<<(NT * INDIM) / 256, 256, 0, stream>>>(W3, w3t);
    k_extract<<<BB, 256, 0, stream>>>(v_src, v_tgt, embs_n, vcur, vtg);

    for (int step = 0; step < NSTEP; ++step) {
        k_mlp12 <<<BB,  256, 0, stream>>>(vcur, vtg, W1, b1, W2, b2, h2b);
        k_gemm3 <<<512, 256, 0, stream>>>(h2b, w3t, b3, logits, step);
        k_argmax<<<BB,  256, 0, stream>>>(logits, gum, step, tsel);
        k_omega <<<BB,  256, 0, stream>>>(coeff, tsel, omg);
        k_rotate<<<BB,   64, 0, stream>>>(omg, vcur);
    }

    k_cvt_bf16<<<(BB * NN + 255) / 256, 256, 0, stream>>>(vcur, vfb, BB * NN);
    k_score  <<<16 * 391, 256, 0, stream>>>(vfb, embs_b, pred);
    k_softmax<<<BB,      1024, 0, stream>>>(pred);
}
